// SelfAttentionBlock_88527865905472
// MI455X (gfx1250) — matrix operand order repaired and checked by reading, not compiled, not run on hardware
//
#include <hip/hip_runtime.h>
#include <hip/hip_bf16.h>

#define C_   256
#define W_   4096
#define B_   4
#define G_   32
#define CPG  8            // C_/G_
#define EPS_ 1e-6f

typedef __attribute__((ext_vector_type(16))) _Float16 v16h;
typedef __attribute__((ext_vector_type(8)))  _Float16 v8h;
typedef __attribute__((ext_vector_type(8)))  float    v8f;

// Assemble a 16-half WMMA operand from two 16-byte loads (global or LDS).
static __device__ __forceinline__ v16h ld_frag(const _Float16* lo, const _Float16* hi) {
  v8h a = *(const v8h*)lo;
  v8h b = *(const v8h*)hi;
  v16h r;
#pragma unroll
  for (int i = 0; i < 8; ++i) { r[i] = a[i]; r[i + 8] = b[i]; }
  return r;
}

// CDNA5 async copy: 16 bytes per lane, global -> LDS, tracked by ASYNCcnt.
static __device__ __forceinline__ void async_ld16(unsigned lds_byte_off,
                                                 const _Float16* g) {
  asm volatile("global_load_async_to_lds_b128 %0, %1, off"
               :: "v"(lds_byte_off), "v"(g)
               : "memory");
}

template <int N>
static __device__ __forceinline__ void wait_async() {
  asm volatile("s_wait_asynccnt %0" :: "i"(N) : "memory");
}

// ---------------- Kernel 0: convert Wq/Wk/Wv/Wo to f16 (contiguous [4][C][C]) --
__global__ void k_wcvt(const float* wq, const float* wk, const float* wv,
                       const float* wo, _Float16* w16) {
  int i = blockIdx.x * 256 + threadIdx.x;          // 0 .. 4*C*C-1
  int m = i >> 16;
  int off = i & 65535;
  const float* src = (m == 0) ? wq : (m == 1) ? wk : (m == 2) ? wv : wo;
  w16[i] = (_Float16)src[off];
}

// ---------------- Kernel 1: GroupNorm statistics -------------------------------
__global__ void k_gnstats(const float* x, float* stats) {
  __shared__ float s1[256];
  __shared__ float s2[256];
  int bg = blockIdx.x;                              // b*32 + g
  int b = bg >> 5, g = bg & 31;
  const float* p = x + (size_t)b * C_ * W_ + (size_t)g * CPG * W_;
  float sum = 0.f, sq = 0.f;
  for (int i = threadIdx.x; i < CPG * W_; i += 256) {
    float v = p[i];
    sum += v; sq += v * v;
  }
  s1[threadIdx.x] = sum; s2[threadIdx.x] = sq;
  __syncthreads();
  for (int s = 128; s > 0; s >>= 1) {
    if (threadIdx.x < (unsigned)s) {
      s1[threadIdx.x] += s1[threadIdx.x + s];
      s2[threadIdx.x] += s2[threadIdx.x + s];
    }
    __syncthreads();
  }
  if (threadIdx.x == 0) {
    float inv_n = 1.0f / (CPG * W_);
    float mean = s1[0] * inv_n;
    float var  = s2[0] * inv_n - mean * mean;
    stats[bg * 2 + 0] = mean;
    stats[bg * 2 + 1] = rsqrtf(var + EPS_);
  }
}

// ---------------- Kernel 2: GN apply -> normT [b][w][c] f16 --------------------
__global__ void k_gnapply(const float* x, const float* gamma, const float* beta,
                          const float* stats, _Float16* normT) {
  int i = blockIdx.x * 256 + threadIdx.x;           // over B*C*W
  int w = i % W_;
  int c = (i / W_) % C_;
  int b = i / (W_ * C_);
  int bg = b * G_ + c / CPG;
  float mean = stats[bg * 2 + 0];
  float rstd = stats[bg * 2 + 1];
  float v = (x[i] - mean) * rstd * gamma[c] + beta[c];
  normT[((size_t)b * W_ + w) * C_ + c] = (_Float16)v;
}

// ---------------- Kernel 3: Q/K/V projections (WMMA GEMMs) ---------------------
__global__ void k_proj(const _Float16* w16, const _Float16* normT,
                       const float* bq, const float* bk, const float* bv,
                       _Float16* q, _Float16* k, _Float16* v) {
  int wid  = threadIdx.x >> 5;
  int lane = threadIdx.x & 31;
  int col  = lane & 15;
  int grp  = lane >> 4;
  int gw = blockIdx.x * 8 + wid;                    // ((which*B+b)*256+wt)*16+ot
  int ot = gw & 15; gw >>= 4;
  int wt = gw & 255; gw >>= 8;
  int b  = gw & 3;
  int which = gw >> 2;                              // 0=q,1=k,2=v
  const _Float16* Wm  = w16 + which * 65536;
  const float* bias = (which == 0) ? bq : (which == 1) ? bk : bv;
  int o_lane = ot * 16 + col;                       // A-operand row (M)
  int w_lane = wt * 16 + col;                       // B-operand col (N)
  const _Float16* pn = normT + ((size_t)b * W_ + w_lane) * C_;
  v8f acc = {};
#pragma unroll
  for (int kc = 0; kc < 8; ++kc) {                  // K = 256 = 8 x 32
    const _Float16* pa = Wm + o_lane * C_ + kc * 32 + grp * 16;
    v16h a = ld_frag(pa, pa + 8);
    const _Float16* pb = pn + kc * 32 + grp * 16;
    v16h bf = ld_frag(pb, pb + 8);
    acc = __builtin_amdgcn_wmma_f32_16x16x32_f16(false, a, false, bf,
                                                 (short)0, acc, false, false);
  }
  int obase = ot * 16 + grp * 8;                    // 8 consecutive output rows
  if (which == 2) {                                 // V -> [b][c][w]
#pragma unroll
    for (int r = 0; r < 8; ++r) {
      float val = acc[r] + bias[obase + r];
      v[((size_t)b * C_ + obase + r) * W_ + w_lane] = (_Float16)val;
    }
  } else {                                          // Q,K -> [b][w][c], Q scaled
    float scale = (which == 0) ? 0.0625f : 1.0f;    // C^-0.5 = 1/16
    v8h out;
#pragma unroll
    for (int r = 0; r < 8; ++r)
      out[r] = (_Float16)((acc[r] + bias[obase + r]) * scale);
    _Float16* dst = ((which == 0) ? q : k) + ((size_t)b * W_ + w_lane) * C_ + obase;
    *(v8h*)dst = out;
  }
}

// ---------------- Kernel 4: flash attention with async LDS staging -------------
// 8 waves per block, one 16-query tile each, same batch.  K tiles (32 keys x
// 256 ch) double-buffered via global_load_async_to_lds_b128; V tile shared.
__global__ void __launch_bounds__(256, 1)
k_attn(const _Float16* q, const _Float16* kk, const _Float16* vv, _Float16* attnT) {
  __shared__ __align__(16) _Float16 kbuf[2][32 * C_];  // [j'][c]   2 x 16 KB
  __shared__ __align__(16) _Float16 vtile[C_ * 32];    // [c][j']       16 KB
  __shared__ __align__(16) _Float16 pbuf[8][16 * 32];  // P [i][j], per wave
  __shared__ float abuf[8][16];
  __shared__ float sbuf[8][16];

  int wid  = threadIdx.x >> 5;
  int lane = threadIdx.x & 31;
  int col  = lane & 15;
  int grp  = lane >> 4;
  int b  = blockIdx.x >> 5;                          // 32 blocks per batch
  int it = (blockIdx.x & 31) * 8 + wid;
  int ibase = it * 16;

  const _Float16* kbB = kk + (size_t)b * W_ * C_;    // [j][c]
  const _Float16* vbA = vv + (size_t)b * C_ * W_;    // [c][j]

  // cache Q A-fragments for this query tile (K=256 -> 8 chunks)
  v16h qf[8];
  const _Float16* pq = q + ((size_t)b * W_ + ibase + col) * C_;
#pragma unroll
  for (int kc = 0; kc < 8; ++kc) {
    const _Float16* p = pq + kc * 32 + grp * 16;
    qf[kc] = ld_frag(p, p + 8);
  }
  float mrun[8], srun[8];
#pragma unroll
  for (int r = 0; r < 8; ++r) { mrun[r] = -3.0e38f; srun[r] = 0.f; }
  v8f zero = {};
  v8f osum[16];
#pragma unroll
  for (int ct = 0; ct < 16; ++ct) osum[ct] = zero;

  unsigned k0off = (unsigned)(size_t)&kbuf[0][0];
  unsigned k1off = (unsigned)(size_t)&kbuf[1][0];
  unsigned voff  = (unsigned)(size_t)&vtile[0];

  // prologue: async-stage K chunk 0 (16 KB contiguous, 4 x b128 per lane)
#pragma unroll
  for (int e = 0; e < 4; ++e) {
    int seg = threadIdx.x + e * 256;                 // 0..1023 16B segments
    async_ld16(k0off + seg * 16, kbB + seg * 8);
  }

  const int NCH = W_ / 32;
  for (int jc = 0; jc < NCH; ++jc) {
    int jbase = jc * 32;
    const _Float16* kt = &kbuf[jc & 1][0];
    unsigned knext = (jc & 1) ? k0off : k1off;
    // issue V(jc) first, then K(jc+1): async loads complete in order per wave
#pragma unroll
    for (int e = 0; e < 4; ++e) {
      int idx = threadIdx.x + e * 256;               // 1024 segs: 256 rows x 4
      int row = idx >> 2, seg = idx & 3;
      async_ld16(voff + (row * 32 + seg * 8) * 2,
                 vbA + (size_t)row * W_ + jbase + seg * 8);
    }
    if (jc + 1 < NCH) {
#pragma unroll
      for (int e = 0; e < 4; ++e) {
        int seg = threadIdx.x + e * 256;
        async_ld16(knext + seg * 16, kbB + (size_t)(jbase + 32) * C_ + seg * 8);
      }
      wait_async<8>();                               // retire K(jc)
    } else {
      wait_async<4>();                               // retire K(jc)
    }
    __syncthreads();                                 // K(jc) visible to all

    // scores: two 16x16 tiles from LDS K, K-dim = 256
    v8f S0 = zero, S1 = zero;
#pragma unroll
    for (int kc = 0; kc < 8; ++kc) {
      const _Float16* pb0 = kt + col * C_ + kc * 32 + grp * 16;
      v16h b0 = ld_frag(pb0, pb0 + 8);
      S0 = __builtin_amdgcn_wmma_f32_16x16x32_f16(false, qf[kc], false, b0,
                                                  (short)0, S0, false, false);
      const _Float16* pb1 = kt + (16 + col) * C_ + kc * 32 + grp * 16;
      v16h b1 = ld_frag(pb1, pb1 + 8);
      S1 = __builtin_amdgcn_wmma_f32_16x16x32_f16(false, qf[kc], false, b1,
                                                  (short)0, S1, false, false);
    }
    // online softmax; lane-half holds rows grp*8 + r, col = N = j%16
#pragma unroll
    for (int r = 0; r < 8; ++r) {
      float t = fmaxf(S0[r], S1[r]);
      t = fmaxf(t, __shfl_xor(t, 1, 32));
      t = fmaxf(t, __shfl_xor(t, 2, 32));
      t = fmaxf(t, __shfl_xor(t, 4, 32));
      t = fmaxf(t, __shfl_xor(t, 8, 32));
      float mnew  = fmaxf(mrun[r], t);
      float alpha = __expf(mrun[r] - mnew);
      float e0 = __expf(S0[r] - mnew);
      float e1 = __expf(S1[r] - mnew);
      float rs = e0 + e1;
      rs += __shfl_xor(rs, 1, 32);
      rs += __shfl_xor(rs, 2, 32);
      rs += __shfl_xor(rs, 4, 32);
      rs += __shfl_xor(rs, 8, 32);
      srun[r] = srun[r] * alpha + rs;
      mrun[r] = mnew;
      int row = grp * 8 + r;
      pbuf[wid][row * 32 + col]      = (_Float16)e0;
      pbuf[wid][row * 32 + 16 + col] = (_Float16)e1;
      if (col == 0) abuf[wid][row] = alpha;
    }
    float aCol = abuf[wid][col];                     // rescale factor per i=col
#pragma unroll
    for (int ct = 0; ct < 16; ++ct)
#pragma unroll
      for (int e = 0; e < 8; ++e) osum[ct][e] *= aCol;

    if (jc + 1 < NCH) wait_async<4>();               // retire V(jc)
    else              wait_async<0>();
    __syncthreads();                                 // V(jc) visible to all

    // P as B-operand: N=i=col, K=j contiguous 16 halfs
    const _Float16* pp = &pbuf[wid][col * 32 + grp * 16];
    v16h pf = ld_frag(pp, pp + 8);
#pragma unroll
    for (int ct = 0; ct < 16; ++ct) {                // O[c-tile][i] += V * P^T
      const _Float16* pv = &vtile[(ct * 16 + col) * 32 + grp * 16];
      v16h a = ld_frag(pv, pv + 8);
      osum[ct] = __builtin_amdgcn_wmma_f32_16x16x32_f16(false, a, false, pf,
                                                        (short)0, osum[ct],
                                                        false, false);
    }
    __syncthreads();                                 // protect vtile reuse
  }
  // finalize: divide by row sums, store attnT [b][i][c]
#pragma unroll
  for (int r = 0; r < 8; ++r)
    if (col == 0) sbuf[wid][grp * 8 + r] = srun[r];
  float inv = 1.0f / sbuf[wid][col];
#pragma unroll
  for (int ct = 0; ct < 16; ++ct) {
    v8h out;
#pragma unroll
    for (int r = 0; r < 8; ++r) out[r] = (_Float16)(osum[ct][r] * inv);
    _Float16* dst = attnT + ((size_t)b * W_ + ibase + col) * C_ + ct * 16 + grp * 8;
    *(v8h*)dst = out;
  }
}

// ---------------- Kernel 5: output projection + bias + residual ----------------
__global__ void k_out(const _Float16* w16, const _Float16* attnT,
                      const float* bo, const float* x, float* out) {
  int wid  = threadIdx.x >> 5;
  int lane = threadIdx.x & 31;
  int col  = lane & 15;
  int grp  = lane >> 4;
  int gw = blockIdx.x * 8 + wid;                    // (b*256+wt)*16+ot
  int ot = gw & 15; gw >>= 4;
  int wt = gw & 255;
  int b  = gw >> 8;
  const _Float16* Wm = w16 + 3 * 65536;             // Wo
  int o_lane = ot * 16 + col;
  int w_lane = wt * 16 + col;
  const _Float16* pn = attnT + ((size_t)b * W_ + w_lane) * C_;
  v8f acc = {};
#pragma unroll
  for (int kc = 0; kc < 8; ++kc) {
    const _Float16* pa = Wm + o_lane * C_ + kc * 32 + grp * 16;
    v16h a = ld_frag(pa, pa + 8);
    const _Float16* pb = pn + kc * 32 + grp * 16;
    v16h bf = ld_frag(pb, pb + 8);
    acc = __builtin_amdgcn_wmma_f32_16x16x32_f16(false, a, false, bf,
                                                 (short)0, acc, false, false);
  }
  int obase = ot * 16 + grp * 8;
#pragma unroll
  for (int r = 0; r < 8; ++r) {
    size_t idx = ((size_t)b * C_ + obase + r) * W_ + w_lane;
    out[idx] = acc[r] + bo[obase + r] + x[idx];
  }
}

// ---------------- Host-side launch ---------------------------------------------
extern "C" void kernel_launch(void* const* d_in, const int* in_sizes, int n_in,
                              void* d_out, int out_size, void* d_ws, size_t ws_size,
                              hipStream_t stream) {
  const float* x     = (const float*)d_in[0];
  const float* gamma = (const float*)d_in[1];
  const float* beta  = (const float*)d_in[2];
  const float* Wq    = (const float*)d_in[3];
  const float* bq    = (const float*)d_in[4];
  const float* Wk    = (const float*)d_in[5];
  const float* bk    = (const float*)d_in[6];
  const float* Wv    = (const float*)d_in[7];
  const float* bv    = (const float*)d_in[8];
  const float* Wo    = (const float*)d_in[9];
  const float* bo    = (const float*)d_in[10];
  float* out = (float*)d_out;

  char* ws = (char*)d_ws;
  const size_t HSZ = (size_t)B_ * W_ * C_ * sizeof(_Float16);   // 8 MB each
  float*    stats = (float*)ws;                                  // 1 KB
  _Float16* w16   = (_Float16*)(ws + 1024);                      // 512 KB
  _Float16* normT = (_Float16*)(ws + 1024 + 4 * 65536 * sizeof(_Float16));
  _Float16* q     = (_Float16*)((char*)normT + HSZ);
  _Float16* k     = (_Float16*)((char*)q + HSZ);
  _Float16* v     = (_Float16*)((char*)k + HSZ);
  _Float16* attnT = (_Float16*)((char*)v + HSZ);

  k_wcvt  <<<(4 * C_ * C_) / 256, 256, 0, stream>>>(Wq, Wk, Wv, Wo, w16);
  k_gnstats<<<B_ * G_, 256, 0, stream>>>(x, stats);
  k_gnapply<<<(B_ * C_ * W_) / 256, 256, 0, stream>>>(x, gamma, beta, stats, normT);
  k_proj  <<<(3 * B_ * (W_ / 16) * (C_ / 16)) / 8, 256, 0, stream>>>(
      w16, normT, bq, bk, bv, q, k, v);
  k_attn  <<<(B_ * (W_ / 16)) / 8, 256, 0, stream>>>(q, k, v, attnT);
  k_out   <<<(B_ * (W_ / 16) * (C_ / 16)) / 8, 256, 0, stream>>>(
      w16, attnT, bo, x, out);
}
